// NeuralCellularAutomata_16415365005781
// MI455X (gfx1250) — matrix operand order repaired and checked by reading, not compiled, not run on hardware
//
#include <hip/hip_runtime.h>

typedef __attribute__((ext_vector_type(16))) _Float16 v16h;
typedef __attribute__((ext_vector_type(8)))  float    v8f;
typedef __attribute__((ext_vector_type(8)))  unsigned v8u;

#define NB    128      // batch
#define NC    64       // state channels
#define NMLP  512      // mlp channels
#define HW    40
#define NPIX  1600     // 40*40
#define K1    192      // 3*NC perception channels
#define NTILE 100      // 10x10 tiles of 4x4 pixels
#define NORMN (NMLP*NPIX)

// ---- workspace layout (bytes) ----
#define OFF_W1H   0                        // 512*192 f16 = 196608 B
#define OFF_W2H   196608                   // 64*512  f16 = 65536 B
#define OFF_PS1   262144                   // 12800 f32 partial sums
#define OFF_PS2   (262144 + 51200)         // 12800 f32 partial sumsq
#define OFF_MU    (262144 + 102400)        // 128 f32
#define OFF_RSTD  (OFF_MU + 512)           // 128 f32

// ---------------------------------------------------------------------------
// prep: cast w1 [512x192] and w2 [64x512] fp32 -> fp16 (row-major; K-pairs
// contiguous so each WMMA A-operand dword is one u32 load).
// ---------------------------------------------------------------------------
__global__ void __launch_bounds__(256) nca_prep(const float* __restrict__ w1,
                                                const float* __restrict__ w2,
                                                _Float16* __restrict__ w1h,
                                                _Float16* __restrict__ w2h) {
  int i = blockIdx.x * 256 + threadIdx.x;
  if (i < NMLP * K1)            w1h[i] = (_Float16)w1[i];
  else if (i < NMLP * K1 + NC * NMLP) {
    int j = i - NMLP * K1;
    w2h[j] = (_Float16)w2[j];
  }
}

// ---------------------------------------------------------------------------
// Perception: depthwise [sx, sy, identity] 3x3 stencils with zero ('SAME')
// padding for a 4x4 pixel tile. Output pix-major into LDS: pbuf[pix*192 + k],
// k = 3*c + f, so B-matrix K-pairs are contiguous (one 32-bit DS load each).
// ---------------------------------------------------------------------------
__device__ __forceinline__ void nca_perception(const float* __restrict__ state,
                                               int b, int y0, int x0,
                                               _Float16* __restrict__ pbuf) {
  for (int v = threadIdx.x; v < 16 * K1; v += 256) {
    int k = v >> 4, pix = v & 15;
    int c = k / 3, f = k - 3 * c;
    int y = y0 + (pix >> 2), x = x0 + (pix & 3);
    const float* sc = state + ((size_t)b * NC + c) * NPIX;
    auto at = [&](int yy, int xx) -> float {
      return (yy >= 0 && yy < HW && xx >= 0 && xx < HW) ? sc[yy * HW + xx] : 0.f;
    };
    float r;
    if (f == 0) {                       // sobel-x
      r = -at(y-1,x-1) + at(y-1,x+1)
        - 2.f*at(y,x-1) + 2.f*at(y,x+1)
        - at(y+1,x-1) + at(y+1,x+1);
    } else if (f == 1) {                // sobel-y
      r = -at(y-1,x-1) - 2.f*at(y-1,x) - at(y-1,x+1)
        + at(y+1,x-1) + 2.f*at(y+1,x) + at(y+1,x+1);
    } else {                            // identity
      r = at(y,x);
    }
    pbuf[pix * K1 + k] = (_Float16)r;
  }
}

// ---------------------------------------------------------------------------
// One 16(o) x 16(pix) conv1 tile: K=192 in 6 chunks of 32 via
// v_wmma_f32_16x16x32_f16.  A = w1h rows (u32 = f16 pair), B = perception LDS.
// A layout (16x32 f16): lane row = lane&15, half = lane>>4,
//   VGPR v: k = kc*32 + (v>>2)*16 + half*8 + (v&3)*2   -> u32 idx kc*16+(v>>2)*8+half*4+(v&3)
// B layout (32x16 f16): lane col = lane&15,
//   VGPR v: k = kc*32 + (v>>2)*16 + half*8 + (v&3)*2   -> u32 idx kc*16+(v>>2)*8+half*4+(v&3)
// ---------------------------------------------------------------------------
__device__ __forceinline__ v8f nca_conv1_tile(const unsigned* __restrict__ w1u,
                                              const unsigned* __restrict__ pu,
                                              int ot, int lane) {
  int row = lane & 15, half = lane >> 4, col = lane & 15;
  const unsigned* arow = w1u + (size_t)(ot * 16 + row) * (K1 / 2);
  const unsigned* bcol = pu + col * (K1 / 2);
  v8f c = {};
#pragma unroll
  for (int kc = 0; kc < 6; ++kc) {
    v8u au, bu;
#pragma unroll
    for (int v = 0; v < 8; ++v) {
      au[v] = arow[kc * 16 + (v >> 2) * 8 + half * 4 + (v & 3)];
      bu[v] = bcol[kc * 16 + (v >> 2) * 8 + half * 4 + (v & 3)];
    }
    c = __builtin_amdgcn_wmma_f32_16x16x32_f16(false, __builtin_bit_cast(v16h, au),
                                               false, __builtin_bit_cast(v16h, bu),
                                               (short)0, c, false, false);
  }
  return c;
}

// ---------------------------------------------------------------------------
// Pass 1: conv1 WMMA per tile, deterministic block tree-reduction of
// sum / sumsq -> per-block partials (no atomics -> replay-deterministic).
// Block = (batch, 4x4 tile); 8 waves x 4 otiles cover all 512 mlp channels.
// ---------------------------------------------------------------------------
__global__ void __launch_bounds__(256) nca_pass1(const float* __restrict__ state,
                                                 const _Float16* __restrict__ w1h,
                                                 float* __restrict__ ps1,
                                                 float* __restrict__ ps2) {
  __shared__ __align__(16) _Float16 pbuf[16 * K1];
  __shared__ float red1[256], red2[256];
  int b = blockIdx.x / NTILE, tile = blockIdx.x % NTILE;
  int y0 = (tile / 10) * 4, x0 = (tile % 10) * 4;
  nca_perception(state, b, y0, x0, pbuf);
  __syncthreads();

  int wave = threadIdx.x >> 5, lane = threadIdx.x & 31;
  const unsigned* w1u = (const unsigned*)w1h;
  const unsigned* pu  = (const unsigned*)pbuf;
  float s1 = 0.f, s2 = 0.f;
  for (int t = 0; t < 4; ++t) {
    v8f c = nca_conv1_tile(w1u, pu, wave * 4 + t, lane);
#pragma unroll
    for (int j = 0; j < 8; ++j) { s1 += c[j]; s2 += c[j] * c[j]; }
  }
  red1[threadIdx.x] = s1; red2[threadIdx.x] = s2;
  __syncthreads();
  for (int s = 128; s > 0; s >>= 1) {
    if (threadIdx.x < s) {
      red1[threadIdx.x] += red1[threadIdx.x + s];
      red2[threadIdx.x] += red2[threadIdx.x + s];
    }
    __syncthreads();
  }
  if (threadIdx.x == 0) { ps1[blockIdx.x] = red1[0]; ps2[blockIdx.x] = red2[0]; }
}

// ---------------------------------------------------------------------------
// Stats: fold 100 tile partials per batch -> mu, rsqrt(var+eps).
// ---------------------------------------------------------------------------
__global__ void nca_stats(const float* __restrict__ ps1, const float* __restrict__ ps2,
                          float* __restrict__ mu, float* __restrict__ rstd) {
  int b = threadIdx.x;
  if (b < NB) {
    float s1 = 0.f, s2 = 0.f;
    for (int t = 0; t < NTILE; ++t) { s1 += ps1[b * NTILE + t]; s2 += ps2[b * NTILE + t]; }
    float m = s1 / (float)NORMN;
    float var = s2 / (float)NORMN - m * m;
    mu[b] = m;
    rstd[b] = rsqrtf(var + 1e-5f);
  }
}

// ---------------------------------------------------------------------------
// Pass 2: recompute conv1 (state/w1 are L2-resident), LN affine + ReLU ->
// f16 LDS (pix-major, K-pairs contiguous), conv2 WMMA, mask*alive gate,
// residual add, store.
// ---------------------------------------------------------------------------
__global__ void __launch_bounds__(256) nca_pass2(const float* __restrict__ state,
                                                 const float* __restrict__ lnw,
                                                 const float* __restrict__ lnb,
                                                 const int*   __restrict__ mask,
                                                 const _Float16* __restrict__ w1h,
                                                 const _Float16* __restrict__ w2h,
                                                 const float* __restrict__ mu,
                                                 const float* __restrict__ rstd,
                                                 float* __restrict__ out) {
  __shared__ __align__(16) _Float16 pbuf[16 * K1];
  __shared__ __align__(16) _Float16 ybuf[16 * NMLP];
  __shared__ float factor[16];
  int b = blockIdx.x / NTILE, tile = blockIdx.x % NTILE;
  int y0 = (tile / 10) * 4, x0 = (tile % 10) * 4;
  nca_perception(state, b, y0, x0, pbuf);
  if (threadIdx.x < 16) {                       // per-pixel gate: mask * alive
    int y = y0 + (threadIdx.x >> 2), x = x0 + (threadIdx.x & 3);
    const float* a = state + ((size_t)b * NC + 3) * NPIX;
    float mv = -3.4e38f;
    for (int dy = -1; dy <= 1; ++dy)
      for (int dx = -1; dx <= 1; ++dx) {
        int yy = y + dy, xx = x + dx;
        if (yy >= 0 && yy < HW && xx >= 0 && xx < HW) mv = fmaxf(mv, a[yy * HW + xx]);
      }
    float live = (mv > 0.1f) ? 1.f : 0.f;
    factor[threadIdx.x] = live * (float)mask[y * HW + x];
  }
  __syncthreads();

  int wave = threadIdx.x >> 5, lane = threadIdx.x & 31;
  int half = lane >> 4, col = lane & 15;
  int gp = (y0 + (col >> 2)) * HW + x0 + (col & 3);
  float m = mu[b], rs = rstd[b];
  const unsigned* w1u = (const unsigned*)w1h;
  const unsigned* pu  = (const unsigned*)pbuf;

  // conv1 + LN + ReLU -> ybuf (f16, pix-major over 512 channels)
  for (int t = 0; t < 4; ++t) {
    int ot = wave * 4 + t;
    v8f c = nca_conv1_tile(w1u, pu, ot, lane);
#pragma unroll
    for (int j = 0; j < 8; ++j) {               // D layout: M = half*8 + j, N = col
      int o = ot * 16 + half * 8 + j;
      size_t li = (size_t)o * NPIX + gp;
      float yn = (c[j] - m) * rs * lnw[li] + lnb[li];
      ybuf[col * NMLP + o] = (_Float16)fmaxf(yn, 0.f);
    }
  }
  __syncthreads();

  // conv2: 64 outputs = 4 otiles, waves 0..3, K=512 in 16 chunks of 32
  if (wave < 4) {
    const unsigned* w2u  = (const unsigned*)w2h;
    const unsigned* yu   = (const unsigned*)ybuf;
    const unsigned* arow = w2u + (size_t)(wave * 16 + (lane & 15)) * (NMLP / 2);
    const unsigned* bcol = yu + col * (NMLP / 2);
    v8f c = {};
#pragma unroll
    for (int kc = 0; kc < 16; ++kc) {
      v8u au, bu;
#pragma unroll
      for (int v = 0; v < 8; ++v) {
        au[v] = arow[kc * 16 + (v >> 2) * 8 + half * 4 + (v & 3)];
        bu[v] = bcol[kc * 16 + (v >> 2) * 8 + half * 4 + (v & 3)];
      }
      c = __builtin_amdgcn_wmma_f32_16x16x32_f16(false, __builtin_bit_cast(v16h, au),
                                                 false, __builtin_bit_cast(v16h, bu),
                                                 (short)0, c, false, false);
    }
    float f = factor[col];
#pragma unroll
    for (int j = 0; j < 8; ++j) {
      int o = wave * 16 + half * 8 + j;
      size_t idx = ((size_t)b * NC + o) * NPIX + gp;
      out[idx] = state[idx] + c[j] * f;
    }
  }
}

extern "C" void kernel_launch(void* const* d_in, const int* in_sizes, int n_in,
                              void* d_out, int out_size, void* d_ws, size_t ws_size,
                              hipStream_t stream) {
  (void)in_sizes; (void)n_in; (void)out_size; (void)ws_size;
  const float* state = (const float*)d_in[0];   // [128,64,40,40]
  const float* w1    = (const float*)d_in[1];   // [512,192]
  const float* lnw   = (const float*)d_in[2];   // [512,40,40]
  const float* lnb   = (const float*)d_in[3];   // [512,40,40]
  const float* w2    = (const float*)d_in[4];   // [64,512]
  const int*   mask  = (const int*)d_in[5];     // [40,40]

  char* ws = (char*)d_ws;
  _Float16* w1h = (_Float16*)(ws + OFF_W1H);
  _Float16* w2h = (_Float16*)(ws + OFF_W2H);
  float* ps1  = (float*)(ws + OFF_PS1);
  float* ps2  = (float*)(ws + OFF_PS2);
  float* mu   = (float*)(ws + OFF_MU);
  float* rstd = (float*)(ws + OFF_RSTD);

  nca_prep<<<512, 256, 0, stream>>>(w1, w2, w1h, w2h);
  nca_pass1<<<NB * NTILE, 256, 0, stream>>>(state, w1h, ps1, ps2);
  nca_stats<<<1, 128, 0, stream>>>(ps1, ps2, mu, rstd);
  nca_pass2<<<NB * NTILE, 256, 0, stream>>>(state, lnw, lnb, mask, w1h, w2h,
                                            mu, rstd, (float*)d_out);
}
